// BasicBlock_22866405884045
// MI455X (gfx1250) — matrix operand order repaired and checked by reading, not compiled, not run on hardware
//
#include <hip/hip_runtime.h>
#include <stdint.h>

// ---------------- constants for this problem ----------------
#define N_PTS   131072
#define C_CH    128
#define KOFF    27
#define E_DIM   512
#define NB      8
#define EPS_F   1e-6f

typedef __attribute__((ext_vector_type(16))) __bf16 v16bf;
typedef __attribute__((ext_vector_type(8)))  __bf16 v8bf;
typedef __attribute__((ext_vector_type(8)))  float  v8f;

union FragBF { v16bf v; v8bf h[2]; };

// ---------------- helpers ----------------
__device__ __forceinline__ uint16_t f2bf(float f) {
  uint32_t u = __float_as_uint(f);
  uint32_t r = u + 0x7FFFu + ((u >> 16) & 1u);
  if ((u & 0x7F800000u) == 0x7F800000u) r = u;  // inf/nan passthrough
  return (uint16_t)(r >> 16);
}

__device__ __forceinline__ float elu_f(float x) {
  return x > 0.0f ? x : (__expf(x) - 1.0f);
}

// Async copy of one 128-byte slice per thread: global -> LDS (ASYNCcnt-tracked).
// Each of 256 threads moves 128 B => 32 KB total (one W_k image).
__device__ __forceinline__ void async_copy_wk(const uint16_t* __restrict__ wk,
                                              uint16_t* lds_dst, int tid) {
  uint64_t g = (uint64_t)(uintptr_t)wk + (uint32_t)tid * 128u;
  uint32_t l = (uint32_t)(uintptr_t)lds_dst + (uint32_t)tid * 128u;  // LDS offset in addr[31:0]
#pragma unroll
  for (int i = 0; i < 8; ++i) {
    asm volatile("global_load_async_to_lds_b128 %0, %1, off"
                 :: "v"(l), "v"(g) : "memory");
    g += 16u;
    l += 16u;
  }
}

__device__ __forceinline__ void wait_async0() {
  asm volatile("s_wait_asynccnt 0x0" ::: "memory");
}

// ---------------- tiny utility kernels ----------------
__global__ void zero_f32_kernel(float* p, int n) {
  int i = blockIdx.x * blockDim.x + threadIdx.x;
  if (i < n) p[i] = 0.0f;
}

__global__ void cvt_x_kernel(const float* __restrict__ x, uint16_t* __restrict__ xb, int n) {
  int i = blockIdx.x * blockDim.x + threadIdx.x;
  if (i < n) xb[i] = f2bf(x[i]);
}

// W[k][cin][cout] (row-major) -> Wb[k][cout][cin]  (bf16, transposed for B frags)
__global__ void cvt_w_kernel(const float* __restrict__ w, uint16_t* __restrict__ wb, int n) {
  int i = blockIdx.x * blockDim.x + threadIdx.x;
  if (i < n) {
    int k    = i / (C_CH * C_CH);
    int rem  = i - k * (C_CH * C_CH);
    int cin  = rem / C_CH;
    int cout = rem - cin * C_CH;
    wb[(size_t)k * C_CH * C_CH + (size_t)cout * C_CH + cin] = f2bf(w[i]);
  }
}

// e[b, j] = sum_d emb[b,d] * Wt[d, j] + bt[j],  b in [0,8), j in [0,256)
__global__ void film_params_kernel(const float* __restrict__ emb, const float* __restrict__ Wt,
                                   const float* __restrict__ bt, float* __restrict__ e) {
  int j = blockIdx.x * blockDim.x + threadIdx.x;
  if (j < NB * 2 * C_CH) {
    int b = j / (2 * C_CH);
    int c = j - b * (2 * C_CH);
    float acc = bt[c];
    const float* er = emb + (size_t)b * E_DIM;
    for (int d = 0; d < E_DIM; ++d) acc += er[d] * Wt[(size_t)d * (2 * C_CH) + c];
    e[j] = acc;
  }
}

__global__ void count_kernel(const int* __restrict__ bidx, float* __restrict__ cnt, int n) {
  int i = blockIdx.x * blockDim.x + threadIdx.x;
  if (i < n) atomicAdd(&cnt[bidx[i]], 1.0f);
}

// ---------------- WMMA sparse conv: out[n,:] = sum_k xb[nbr[n,k],:] @ Wb[k] ----------------
// block = 256 threads (8 waves). Each wave owns a 16-row x 128-col tile.
// W_k slices (32 KB bf16 each) are double-buffered in LDS via async global->LDS copies.
__global__ __launch_bounds__(256) void sparse_conv_wmma_kernel(
    const uint16_t* __restrict__ xb,   // [N, 128] bf16
    const uint16_t* __restrict__ wb,   // [27][cout=128][cin=128] bf16 (transposed)
    const int*      __restrict__ nbr,  // [N, 27]
    float*          __restrict__ out)  // [N, 128] f32
{
  __shared__ uint16_t smem[2][C_CH * C_CH];   // 2 x 32 KB double buffer

  const int tid  = threadIdx.x;
  const int lane = tid & 31;
  const int wave = tid >> 5;
  const int r0   = blockIdx.x * 128 + wave * 16;   // first row of this wave's tile
  const int m    = lane & 15;                      // row within tile this lane gathers
  const int sel  = lane >> 4;                      // 0: K 0-7,16-23 of chunk; 1: K 8-15,24-31

  v8f acc[8];
#pragma unroll
  for (int j = 0; j < 8; ++j) acc[j] = (v8f)(0.0f);

  // Prefetch W_0 into buffer 0.
  async_copy_wk(wb, &smem[0][0], tid);

  for (int k = 0; k < KOFF; ++k) {
    wait_async0();          // this wave's slice of W_k has landed in LDS
    __syncthreads();        // all waves' slices have landed

    if (k + 1 < KOFF)       // overlap: fetch W_{k+1} into the other buffer
      async_copy_wk(wb + (size_t)(k + 1) * C_CH * C_CH, &smem[(k + 1) & 1][0], tid);

    const uint16_t* wk = &smem[k & 1][0];
    const int idx = nbr[(size_t)(r0 + m) * KOFF + k];
    const uint16_t* xrow = xb + (size_t)idx * C_CH;

#pragma unroll
    for (int c = 0; c < 4; ++c) {                  // Cin chunks of 32
      const int c0 = c * 32;
      // A fragment: 16-bit A 16x32 layout.
      // lanes 0-15: K = c0+[0..7] then c0+16+[0..7]; lanes 16-31: +8.
      FragBF a;
      a.h[0] = *(const v8bf*)(xrow + c0 +      sel * 8);
      a.h[1] = *(const v8bf*)(xrow + c0 + 16 + sel * 8);

#pragma unroll
      for (int j = 0; j < 8; ++j) {                // Cout tiles of 16
        // B fragment from LDS: column n = lane%16 -> cout = j*16 + m; K = c0+8*sel+[0..7], then +16.
        const uint16_t* wrow = wk + (j * 16 + m) * C_CH + c0 + sel * 8;
        FragBF b;
        b.h[0] = *(const v8bf*)(wrow);
        b.h[1] = *(const v8bf*)(wrow + 16);
        acc[j] = __builtin_amdgcn_wmma_f32_16x16x32_bf16(
            /*neg_a=*/false, a.v, /*neg_b=*/false, b.v,
            /*c_mod=*/(short)0, acc[j], /*reuse_a=*/false, /*reuse_b=*/false);
      }
    }
    __syncthreads();        // everyone done reading smem[k&1] before it is refilled (iter k+1's target is the other buffer; iter k+2 reuses this one)
  }

  // Store: VGPR r of acc[j] -> row r0 + r + 8*sel, col j*16 + (lane%16)
  const int rbase = r0 + 8 * sel;
#pragma unroll
  for (int j = 0; j < 8; ++j) {
    const int col = j * 16 + m;
#pragma unroll
    for (int r = 0; r < 8; ++r)
      out[(size_t)(rbase + r) * C_CH + col] = acc[j][r];
  }
}

// ---------------- per-(batch,channel) sum / sumsq with run-compressed atomics ----------------
#define STAT_ROWS 1024
__global__ __launch_bounds__(128) void stats_kernel(const float* __restrict__ v,
                                                    const int* __restrict__ bidx,
                                                    float* __restrict__ sum,
                                                    float* __restrict__ sumsq) {
  const int c  = threadIdx.x;                 // channel
  const int r0 = blockIdx.x * STAT_ROWS;
  int   cur = bidx[r0];
  float s = 0.0f, s2 = 0.0f;
  for (int r = r0; r < r0 + STAT_ROWS; ++r) {
    const int b = bidx[r];
    if (b != cur) {
      atomicAdd(&sum[cur * C_CH + c], s);
      atomicAdd(&sumsq[cur * C_CH + c], s2);
      s = 0.0f; s2 = 0.0f; cur = b;
    }
    const float x = v[(size_t)r * C_CH + c];
    s += x; s2 += x * x;
  }
  atomicAdd(&sum[cur * C_CH + c], s);
  atomicAdd(&sumsq[cur * C_CH + c], s2);
}

// ---------------- norm + FiLM + ELU -> bf16 ----------------
__global__ void norm_film_elu_kernel(const float* __restrict__ v,
                                     const int* __restrict__ bidx,
                                     const float* __restrict__ sum,
                                     const float* __restrict__ sumsq,
                                     const float* __restrict__ cnt,
                                     const float* __restrict__ e,
                                     uint16_t* __restrict__ yb) {
  const int i = blockIdx.x * blockDim.x + threadIdx.x;
  if (i >= N_PTS * C_CH) return;
  const int row = i >> 7, c = i & 127;
  const int b = bidx[row];
  const float inv = 1.0f / fmaxf(cnt[b], 1.0f);
  const float mean = sum[b * C_CH + c] * inv;
  const float var  = fmaxf(sumsq[b * C_CH + c] * inv - mean * mean, 0.0f);
  float xn = (v[i] - mean) * __frsqrt_rn(var + EPS_F);
  const float scale = e[b * 2 * C_CH + c];
  const float shift = e[b * 2 * C_CH + C_CH + c];
  xn = xn * (1.0f + scale) + shift;
  yb[i] = f2bf(elu_f(xn));
}

// ---------------- final: norm + residual + ELU -> f32 out ----------------
__global__ void norm_res_elu_kernel(const float* __restrict__ v,
                                    const float* __restrict__ xres,
                                    const int* __restrict__ bidx,
                                    const float* __restrict__ sum,
                                    const float* __restrict__ sumsq,
                                    const float* __restrict__ cnt,
                                    float* __restrict__ outp) {
  const int i = blockIdx.x * blockDim.x + threadIdx.x;
  if (i >= N_PTS * C_CH) return;
  const int row = i >> 7, c = i & 127;
  const int b = bidx[row];
  const float inv = 1.0f / fmaxf(cnt[b], 1.0f);
  const float mean = sum[b * C_CH + c] * inv;
  const float var  = fmaxf(sumsq[b * C_CH + c] * inv - mean * mean, 0.0f);
  const float xn = (v[i] - mean) * __frsqrt_rn(var + EPS_F);
  outp[i] = elu_f(xn + xres[i]);
}

// ---------------- launcher ----------------
extern "C" void kernel_launch(void* const* d_in, const int* in_sizes, int n_in,
                              void* d_out, int out_size, void* d_ws, size_t ws_size,
                              hipStream_t stream) {
  (void)in_sizes; (void)n_in; (void)out_size; (void)ws_size;
  const float* x    = (const float*)d_in[0];
  const float* emb  = (const float*)d_in[1];
  const float* W1   = (const float*)d_in[2];
  const float* W2   = (const float*)d_in[3];
  const float* Wt   = (const float*)d_in[4];
  const float* bt   = (const float*)d_in[5];
  const int*   bidx = (const int*)d_in[6];
  const int*   nbr  = (const int*)d_in[7];
  float* outp = (float*)d_out;

  // workspace layout (bytes, 256-aligned)
  char* ws = (char*)d_ws;
  uint16_t* xb    = (uint16_t*)(ws + 0);                         // 32 MB
  uint16_t* yb    = (uint16_t*)(ws + 33554432);                  // 32 MB
  uint16_t* w1b   = (uint16_t*)(ws + 67108864);                  // 864 KB
  uint16_t* w2b   = (uint16_t*)(ws + 67993600);                  // 864 KB
  float*    cbuf  = (float*)   (ws + 68878336);                  // 64 MB conv out
  float*    sum   = (float*)   (ws + 135987200);                 // 4 KB
  float*    sumsq = (float*)   (ws + 135991296);                 // 4 KB
  float*    cnt   = (float*)   (ws + 135995392);                 // 32 B (padded)
  float*    efilm = (float*)   (ws + 135995648);                 // 8 KB

  const int NW = KOFF * C_CH * C_CH;       // 442368
  const int NE = N_PTS * C_CH;             // 16777216

  // stage 0: zero stats, build bf16 operands, FiLM params, counts
  zero_f32_kernel<<<(2 * NB * C_CH + 8 + 255) / 256, 256, 0, stream>>>(sum, 2 * NB * C_CH + 8);
  cvt_x_kernel<<<NE / 256, 256, 0, stream>>>(x, xb, NE);
  cvt_w_kernel<<<(NW + 255) / 256, 256, 0, stream>>>(W1, w1b, NW);
  cvt_w_kernel<<<(NW + 255) / 256, 256, 0, stream>>>(W2, w2b, NW);
  film_params_kernel<<<(NB * 2 * C_CH + 255) / 256, 256, 0, stream>>>(emb, Wt, bt, efilm);
  count_kernel<<<N_PTS / 256, 256, 0, stream>>>(bidx, cnt, N_PTS);

  // conv1 -> stats -> norm+FiLM+ELU (bf16)
  sparse_conv_wmma_kernel<<<N_PTS / 128, 256, 0, stream>>>(xb, w1b, nbr, cbuf);
  stats_kernel<<<N_PTS / STAT_ROWS, 128, 0, stream>>>(cbuf, bidx, sum, sumsq);
  norm_film_elu_kernel<<<NE / 256, 256, 0, stream>>>(cbuf, bidx, sum, sumsq, cnt, efilm, yb);

  // conv2 -> stats -> norm+residual+ELU (f32 out)
  zero_f32_kernel<<<(2 * NB * C_CH + 255) / 256, 256, 0, stream>>>(sum, 2 * NB * C_CH);
  sparse_conv_wmma_kernel<<<N_PTS / 128, 256, 0, stream>>>(yb, w2b, nbr, cbuf);
  stats_kernel<<<N_PTS / STAT_ROWS, 128, 0, stream>>>(cbuf, bidx, sum, sumsq);
  norm_res_elu_kernel<<<NE / 256, 256, 0, stream>>>(cbuf, x, bidx, sum, sumsq, cnt, outp);
}
